// CrossAttentionFusion_54631984005330
// MI455X (gfx1250) — compile-verified, matrix operand order checked by reading, not run on hardware
//
#include <hip/hip_runtime.h>
#include <hip/hip_bf16.h>

// ---------------- problem constants ----------------
#define HID   256
#define HEADS 4
#define DH    64
#define KNN   5
#define NLAYERS 3
#define IMG_H 160
#define IMG_W 160
#define BATCH 4
#define NPER  8192
#define NPTS  (BATCH * NPER)     // 32768 points
#define NKV   (NPTS * KNN)       // 163840 key/value rows
#define IMGP  (IMG_H * IMG_W)    // 25600 pixels

typedef __attribute__((ext_vector_type(16))) __bf16 v16bf;
typedef __attribute__((ext_vector_type(8)))  __bf16 v8bf;
typedef __attribute__((ext_vector_type(4)))  __bf16 v4bf;
typedef __attribute__((ext_vector_type(8)))  float  v8f;

// Load a 16-bit WMMA A/B fragment for one K-chunk of 32.
// Per CDNA5 ISA layout: lanes 0-15 hold K=[0..7]+[16..23], lanes 16-31 hold
// K=[8..15]+[24..31] of their row/col -> two 16-byte loads.
__device__ __forceinline__ v16bf load_frag(const __bf16* row, int kofs) {
    union { v16bf v; v8bf h[2]; } u;
    u.h[0] = *(const v8bf*)(row + kofs);
    u.h[1] = *(const v8bf*)(row + kofs + 16);
    return u.v;
}

// ---------------- fp32 -> bf16 conversion ----------------
__global__ void cvt_f32_to_bf16(const float* __restrict__ s, __bf16* __restrict__ d, int n) {
    int i = (blockIdx.x * blockDim.x + threadIdx.x) * 4;
    if (i + 3 < n) {
        float4 f = *(const float4*)(s + i);
        v4bf o;
        o[0] = (__bf16)f.x; o[1] = (__bf16)f.y; o[2] = (__bf16)f.z; o[3] = (__bf16)f.w;
        *(v4bf*)(d + i) = o;
    }
}

// ---------------- image transpose [B,C,P] -> [B,P,C] bf16 ----------------
__global__ void transpose_chw_hwc(const float* __restrict__ img, __bf16* __restrict__ out) {
    __shared__ __bf16 tile[32][33];
    int b  = blockIdx.z;
    int p0 = blockIdx.x * 32;
    int c0 = blockIdx.y * 32;
    int tx = threadIdx.x & 31;
    int ty = threadIdx.x >> 5;                  // 0..7
    const float* src = img + (size_t)b * HID * IMGP;
    __bf16*      dst = out + (size_t)b * IMGP * HID;
#pragma unroll
    for (int i = 0; i < 4; ++i) {
        int c = c0 + ty + i * 8;
        tile[ty + i * 8][tx] = (__bf16)src[(size_t)c * IMGP + p0 + tx];
    }
    __syncthreads();
#pragma unroll
    for (int i = 0; i < 4; ++i) {
        int p = p0 + ty + i * 8;
        dst[(size_t)p * HID + c0 + tx] = tile[tx][ty + i * 8];
    }
}

// ---------------- 5-neighbor bilinear sampling (coalesced over C) ----------------
__global__ void sample_kernel(const float* __restrict__ uv, const __bf16* __restrict__ imgT,
                              __bf16* __restrict__ imgf) {
    int m = blockIdx.x;            // point index 0..NPTS-1
    int c = threadIdx.x;           // channel 0..255
    int b = m / NPER;
    float x = uv[m * 2 + 0] * (float)(IMG_W - 1);
    float y = uv[m * 2 + 1] * (float)(IMG_H - 1);
    const float ox[KNN] = {0.f, 1.f, -1.f, 0.f, 0.f};
    const float oy[KNN] = {0.f, 0.f, 0.f, 1.f, -1.f};
    const __bf16* base = imgT + (size_t)b * IMGP * HID;
#pragma unroll
    for (int j = 0; j < KNN; ++j) {
        float xs = fminf(fmaxf(x + ox[j], 0.f), (float)(IMG_W - 1));
        float ys = fminf(fmaxf(y + oy[j], 0.f), (float)(IMG_H - 1));
        float x0f = floorf(xs), y0f = floorf(ys);
        int x0 = (int)x0f, y0 = (int)y0f;
        int x1 = min(x0 + 1, IMG_W - 1), y1 = min(y0 + 1, IMG_H - 1);
        float wx = xs - x0f, wy = ys - y0f;
        float v00 = (float)base[((size_t)(y0 * IMG_W + x0)) * HID + c];
        float v01 = (float)base[((size_t)(y0 * IMG_W + x1)) * HID + c];
        float v10 = (float)base[((size_t)(y1 * IMG_W + x0)) * HID + c];
        float v11 = (float)base[((size_t)(y1 * IMG_W + x1)) * HID + c];
        float r = v00 * (1.f - wx) * (1.f - wy) + v01 * wx * (1.f - wy)
                + v10 * (1.f - wx) * wy        + v11 * wx * wy;
        imgf[((size_t)m * KNN + j) * HID + c] = (__bf16)r;
    }
}

// ---------------- generic bf16 WMMA GEMM: out = A[rows,K] @ W[N,K]^T + b ----------------
// WG = 4 waves; tile = 64 rows x 256 cols. Each wave: 4 m-tiles x 4 n-tiles.
// Per K-chunk: 4 A-frags + 4 B-frags feed 16 WMMAs (4x reuse of each fragment).
// grid.x = rows/64, grid.y = N/256.
enum { EP_BF16 = 0, EP_XINIT = 1, EP_RELU = 2, EP_RESID = 3, EP_RESID_MASK = 4 };

template <int MODE>
__global__ void __launch_bounds__(128)
gemm_bf16_wmma(const __bf16* __restrict__ A, const __bf16* __restrict__ W,
               const float* __restrict__ bias, const float* __restrict__ resid,
               const unsigned char* __restrict__ mask,
               __bf16* __restrict__ out_bf, float* __restrict__ out_f32,
               int K, int N) {
    const int tid   = threadIdx.x;
    const int wave  = tid >> 5;
    const int lane  = tid & 31;
    const int row0  = blockIdx.x * 64;
    const int col0  = blockIdx.y * 256 + wave * 64;
    const int kofs  = (lane & 16) ? 8 : 0;
    const int nlane = lane & 15;
    const int ghalf = (lane >> 4) << 3;         // 0 or 8

    v8f acc[4][4] = {};
    const int nk = K >> 5;
    for (int kc = 0; kc < nk; ++kc) {
        v16bf a[4], b[4];
#pragma unroll
        for (int mt = 0; mt < 4; ++mt)
            a[mt] = load_frag(A + (size_t)(row0 + mt * 16 + nlane) * K + kc * 32, kofs);
#pragma unroll
        for (int nt = 0; nt < 4; ++nt)
            b[nt] = load_frag(W + (size_t)(col0 + nt * 16 + nlane) * K + kc * 32, kofs);
#pragma unroll
        for (int mt = 0; mt < 4; ++mt)
#pragma unroll
            for (int nt = 0; nt < 4; ++nt)
                acc[mt][nt] = __builtin_amdgcn_wmma_f32_16x16x32_bf16(
                    false, a[mt], false, b[nt], (short)0, acc[mt][nt], false, false);
    }
#pragma unroll
    for (int nt = 0; nt < 4; ++nt) {
        int col = col0 + nt * 16 + nlane;
        float bv = bias[col];
#pragma unroll
        for (int mt = 0; mt < 4; ++mt) {
#pragma unroll
            for (int g = 0; g < 8; ++g) {
                int row = row0 + mt * 16 + g + ghalf;
                float v = acc[mt][nt][g] + bv;
                if (MODE == EP_RELU) v = fmaxf(v, 0.f);
                if (MODE == EP_RESID || MODE == EP_RESID_MASK) v += resid[(size_t)row * HID + col];
                if (MODE == EP_RESID_MASK) v *= mask[row] ? 1.0f : 0.0f;
                if (MODE == EP_BF16 || MODE == EP_XINIT || MODE == EP_RELU || MODE == EP_RESID)
                    out_bf[(size_t)row * N + col] = (__bf16)v;
                if (MODE == EP_XINIT || MODE == EP_RESID || MODE == EP_RESID_MASK)
                    out_f32[(size_t)row * N + col] = v;
            }
        }
    }
}

// ---------------- fused attention layer ----------------
// One WG (4 waves) handles 16 points: q = x@wq^T, k = keys@wk^T, v = vals@wv^T
// (all WMMA into LDS), softmax over 5 keys/head, ao@wo^T + bias + residual,
// LayerNorm, write x (f32 + bf16) in place. Each WG owns its rows exclusively.
__global__ void __launch_bounds__(128)
attn_layer_fused(__bf16* __restrict__ xbf, float* __restrict__ x32,
                 const __bf16* __restrict__ keys, const __bf16* __restrict__ vals,
                 const __bf16* __restrict__ wq, const float* __restrict__ bq,
                 const __bf16* __restrict__ wk, const float* __restrict__ bk,
                 const __bf16* __restrict__ wv, const float* __restrict__ bv,
                 const __bf16* __restrict__ wo, const float* __restrict__ bo,
                 const float* __restrict__ ln_g, const float* __restrict__ ln_b) {
    __shared__ float  q_s[16 * HID];            // 16 KB
    __shared__ __bf16 k_s[16 * KNN * HID];      // 40 KB
    __shared__ __bf16 v_s[16 * KNN * HID];      // 40 KB
    __shared__ float  s_s[16 * HEADS * KNN];    // scores / attn weights
    __shared__ __bf16 ao_s[16 * HID];           // 8 KB
    __shared__ float  o_s[16 * HID];            // 16 KB
    __shared__ float  mu_s[16], rs_s[16];

    const int tid   = threadIdx.x;
    const int wave  = tid >> 5;
    const int lane  = tid & 31;
    const int p0    = blockIdx.x * 16;
    const int kofs  = (lane & 16) ? 8 : 0;
    const int nlane = lane & 15;
    const int ghalf = (lane >> 4) << 3;

    // --- P1a: q = x[p0..p0+15] @ wq^T + bq -> q_s (f32)
    {
        const __bf16* Arow = xbf + (size_t)(p0 + nlane) * HID;
        v8f acc[4] = {};
        for (int kc = 0; kc < 8; ++kc) {
            v16bf a = load_frag(Arow + kc * 32, kofs);
#pragma unroll
            for (int i = 0; i < 4; ++i) {
                v16bf b = load_frag(wq + (size_t)(wave * 64 + i * 16 + nlane) * HID + kc * 32, kofs);
                acc[i] = __builtin_amdgcn_wmma_f32_16x16x32_bf16(false, a, false, b,
                                                                 (short)0, acc[i], false, false);
            }
        }
#pragma unroll
        for (int i = 0; i < 4; ++i) {
            int col = wave * 64 + i * 16 + nlane;
            float bb = bq[col];
#pragma unroll
            for (int g = 0; g < 8; ++g) q_s[(g + ghalf) * HID + col] = acc[i][g] + bb;
        }
    }
    // --- P1b/P1c: k and v tiles. Jobs = 5 m-tiles x 4 n-groups; each job:
    // one A-frag stream reused across 4 n-tiles (4 WMMAs per A fragment).
    for (int jj = wave; jj < 20; jj += 4) {
        int mt = jj >> 2, ng = (jj & 3) * 4;
        const __bf16* Arow = keys + (size_t)(p0 * KNN + mt * 16 + nlane) * HID;
        v8f acc[4] = {};
        for (int kc = 0; kc < 8; ++kc) {
            v16bf a = load_frag(Arow + kc * 32, kofs);
#pragma unroll
            for (int t = 0; t < 4; ++t) {
                v16bf b = load_frag(wk + (size_t)((ng + t) * 16 + nlane) * HID + kc * 32, kofs);
                acc[t] = __builtin_amdgcn_wmma_f32_16x16x32_bf16(false, a, false, b,
                                                                 (short)0, acc[t], false, false);
            }
        }
#pragma unroll
        for (int t = 0; t < 4; ++t) {
            float bb = bk[(ng + t) * 16 + nlane];
#pragma unroll
            for (int g = 0; g < 8; ++g)
                k_s[(mt * 16 + g + ghalf) * HID + (ng + t) * 16 + nlane] = (__bf16)(acc[t][g] + bb);
        }
    }
    for (int jj = wave; jj < 20; jj += 4) {
        int mt = jj >> 2, ng = (jj & 3) * 4;
        const __bf16* Arow = vals + (size_t)(p0 * KNN + mt * 16 + nlane) * HID;
        v8f acc[4] = {};
        for (int kc = 0; kc < 8; ++kc) {
            v16bf a = load_frag(Arow + kc * 32, kofs);
#pragma unroll
            for (int t = 0; t < 4; ++t) {
                v16bf b = load_frag(wv + (size_t)((ng + t) * 16 + nlane) * HID + kc * 32, kofs);
                acc[t] = __builtin_amdgcn_wmma_f32_16x16x32_bf16(false, a, false, b,
                                                                 (short)0, acc[t], false, false);
            }
        }
#pragma unroll
        for (int t = 0; t < 4; ++t) {
            float bb = bv[(ng + t) * 16 + nlane];
#pragma unroll
            for (int g = 0; g < 8; ++g)
                v_s[(mt * 16 + g + ghalf) * HID + (ng + t) * 16 + nlane] = (__bf16)(acc[t][g] + bb);
        }
    }
    __syncthreads();

    // --- P2: scores[m,h,j] = (q . k) * 1/sqrt(DH)
    for (int t = tid; t < 16 * HEADS * KNN; t += 128) {
        int m = t / (HEADS * KNN);
        int r = t % (HEADS * KNN);
        int h = r / KNN, j = r % KNN;
        const float*  qp = q_s + m * HID + h * DH;
        const __bf16* kp = k_s + (m * KNN + j) * HID + h * DH;
        float dot = 0.f;
        for (int d = 0; d < DH; ++d) dot += qp[d] * (float)kp[d];
        s_s[t] = dot * 0.125f;
    }
    __syncthreads();

    // --- P3: softmax over KNN (one (m,h) per thread)
    if (tid < 16 * HEADS) {
        int base = tid * KNN;
        float mx = -1e30f;
#pragma unroll
        for (int j = 0; j < KNN; ++j) mx = fmaxf(mx, s_s[base + j]);
        float e[KNN], sum = 0.f;
#pragma unroll
        for (int j = 0; j < KNN; ++j) { e[j] = __expf(s_s[base + j] - mx); sum += e[j]; }
        float inv = 1.f / sum;
#pragma unroll
        for (int j = 0; j < KNN; ++j) s_s[base + j] = e[j] * inv;
    }
    __syncthreads();

    // --- P4: ao = attn @ v
    for (int t = tid; t < 16 * HID; t += 128) {
        int m = t >> 8, c = t & 255, h = c >> 6;
        const float* ap = s_s + (m * HEADS + h) * KNN;
        float s = 0.f;
#pragma unroll
        for (int j = 0; j < KNN; ++j) s += ap[j] * (float)v_s[(m * KNN + j) * HID + c];
        ao_s[t] = (__bf16)s;
    }
    __syncthreads();

    // --- P5: o = ao @ wo^T + bo + x(resid) -> o_s
    {
        const __bf16* Arow = ao_s + nlane * HID;
        v8f acc[4] = {};
        for (int kc = 0; kc < 8; ++kc) {
            v16bf a = load_frag(Arow + kc * 32, kofs);
#pragma unroll
            for (int i = 0; i < 4; ++i) {
                v16bf b = load_frag(wo + (size_t)(wave * 64 + i * 16 + nlane) * HID + kc * 32, kofs);
                acc[i] = __builtin_amdgcn_wmma_f32_16x16x32_bf16(false, a, false, b,
                                                                 (short)0, acc[i], false, false);
            }
        }
#pragma unroll
        for (int i = 0; i < 4; ++i) {
            int col = wave * 64 + i * 16 + nlane;
            float bb = bo[col];
#pragma unroll
            for (int g = 0; g < 8; ++g) {
                int row = p0 + g + ghalf;
                o_s[(g + ghalf) * HID + col] = acc[i][g] + bb + x32[(size_t)row * HID + col];
            }
        }
    }
    __syncthreads();

    // --- P6: LayerNorm stats per row
    if (tid < 16) {
        float s = 0.f;
        for (int c = 0; c < HID; ++c) s += o_s[tid * HID + c];
        float mu = s * (1.f / HID);
        float var = 0.f;
        for (int c = 0; c < HID; ++c) { float d = o_s[tid * HID + c] - mu; var += d * d; }
        mu_s[tid] = mu;
        rs_s[tid] = rsqrtf(var * (1.f / HID) + 1e-5f);
    }
    __syncthreads();

    // --- P7: normalize + write x (f32 + bf16) in place
    for (int t = tid; t < 16 * HID; t += 128) {
        int m = t >> 8, c = t & 255;
        float xv = (o_s[t] - mu_s[m]) * rs_s[m] * ln_g[c] + ln_b[c];
        x32[(size_t)(p0 + m) * HID + c] = xv;
        xbf[(size_t)(p0 + m) * HID + c] = (__bf16)xv;
    }
}

// ---------------- host launcher ----------------
// bf16 weight pool element offsets
#define QP_OFF 0
#define KP_OFF 65536
#define VP_OFF 131072
#define LBASE_OFF 196608
#define LSTRIDE_OFF 786432
#define LWQ 0
#define LWK 65536
#define LWV 131072
#define LWO 196608
#define LF0 262144
#define LF1 524288

// workspace byte offsets (all 1 MiB aligned)
static const size_t WB_OFF   = 0;                      // bf16 weights, ~4.9 MiB
static const size_t IMGT_OFF = 6291456;                // [B,P,C] bf16, 50 MiB
static const size_t IMGF_OFF = 58720256;               // [NKV,256] bf16, 80 MiB
static const size_t KEYS_OFF = 142606336;              // 80 MiB
static const size_t VALS_OFF = 226492416;              // 80 MiB
static const size_t X32_OFF  = 310378496;              // 32 MiB
static const size_t XBF_OFF  = 343932928;              // 16 MiB
static const size_t PFBF_OFF = IMGT_OFF;               // reuse (imgT dead after sampling)
static const size_t H_OFF    = IMGF_OFF;               // reuse (imgf dead after k/v proj)

extern "C" void kernel_launch(void* const* d_in, const int* in_sizes, int n_in,
                              void* d_out, int out_size, void* d_ws, size_t ws_size,
                              hipStream_t stream) {
    (void)in_sizes; (void)n_in; (void)out_size; (void)ws_size;
    const float* pf   = (const float*)d_in[0];
    const float* img  = (const float*)d_in[1];
    const float* uv   = (const float*)d_in[3];
    const unsigned char* mask = (const unsigned char*)d_in[6];
    const float* qp_w = (const float*)d_in[7];
    const float* qp_b = (const float*)d_in[8];
    const float* kp_w = (const float*)d_in[9];
    const float* kp_b = (const float*)d_in[10];
    const float* vp_w = (const float*)d_in[11];
    const float* vp_b = (const float*)d_in[12];
    // layers start at 17; 14 tensors per layer:
    // wq.w wq.b wk.w wk.b wv.w wv.b wo.w wo.b ln.g ln.be ffn0.w ffn0.b ffn1.w ffn1.b

    char* ws = (char*)d_ws;
    __bf16* Wb    = (__bf16*)(ws + WB_OFF);
    __bf16* imgT  = (__bf16*)(ws + IMGT_OFF);
    __bf16* imgf  = (__bf16*)(ws + IMGF_OFF);
    __bf16* keysb = (__bf16*)(ws + KEYS_OFF);
    __bf16* valsb = (__bf16*)(ws + VALS_OFF);
    float*  x32   = (float*) (ws + X32_OFF);
    __bf16* xbf   = (__bf16*)(ws + XBF_OFF);
    __bf16* pfbf  = (__bf16*)(ws + PFBF_OFF);
    __bf16* hbuf  = (__bf16*)(ws + H_OFF);

    auto cvt = [&](const float* src, __bf16* dst, int n) {
        cvt_f32_to_bf16<<<dim3(n / 1024), dim3(256), 0, stream>>>(src, dst, n);
    };

    // 1) convert weights to bf16 pool
    cvt(qp_w, Wb + QP_OFF, 65536);
    cvt(kp_w, Wb + KP_OFF, 65536);
    cvt(vp_w, Wb + VP_OFF, 65536);
    for (int l = 0; l < NLAYERS; ++l) {
        const int di = 17 + l * 14;
        size_t lb = LBASE_OFF + (size_t)l * LSTRIDE_OFF;
        cvt((const float*)d_in[di + 0],  Wb + lb + LWQ, 65536);
        cvt((const float*)d_in[di + 2],  Wb + lb + LWK, 65536);
        cvt((const float*)d_in[di + 4],  Wb + lb + LWV, 65536);
        cvt((const float*)d_in[di + 6],  Wb + lb + LWO, 65536);
        cvt((const float*)d_in[di + 10], Wb + lb + LF0, 262144);
        cvt((const float*)d_in[di + 12], Wb + lb + LF1, 262144);
    }

    // 2) image transpose + 5-neighbor bilinear sampling
    transpose_chw_hwc<<<dim3(IMGP / 32, HID / 32, BATCH), dim3(256), 0, stream>>>(img, imgT);
    sample_kernel<<<dim3(NPTS), dim3(256), 0, stream>>>(uv, imgT, imgf);

    // 3) keys / values projections (bf16 out)
    gemm_bf16_wmma<EP_BF16><<<dim3(NKV / 64, 1), dim3(128), 0, stream>>>(
        imgf, Wb + KP_OFF, kp_b, nullptr, nullptr, keysb, nullptr, 256, 256);
    gemm_bf16_wmma<EP_BF16><<<dim3(NKV / 64, 1), dim3(128), 0, stream>>>(
        imgf, Wb + VP_OFF, vp_b, nullptr, nullptr, valsb, nullptr, 256, 256);

    // 4) x = query_proj(point_features)   (imgT region is dead now -> pfbf)
    cvt(pf, pfbf, NPTS * HID);
    gemm_bf16_wmma<EP_XINIT><<<dim3(NPTS / 64, 1), dim3(128), 0, stream>>>(
        pfbf, Wb + QP_OFF, qp_b, nullptr, nullptr, xbf, x32, 256, 256);

    // 5) transformer layers
    for (int l = 0; l < NLAYERS; ++l) {
        const int di = 17 + l * 14;
        size_t lb = LBASE_OFF + (size_t)l * LSTRIDE_OFF;
        attn_layer_fused<<<dim3(NPTS / 16), dim3(128), 0, stream>>>(
            xbf, x32, keysb, valsb,
            Wb + lb + LWQ, (const float*)d_in[di + 1],
            Wb + lb + LWK, (const float*)d_in[di + 3],
            Wb + lb + LWV, (const float*)d_in[di + 5],
            Wb + lb + LWO, (const float*)d_in[di + 7],
            (const float*)d_in[di + 8], (const float*)d_in[di + 9]);
        // h = relu(ffn0(x))   (imgf region is dead -> hbuf)
        gemm_bf16_wmma<EP_RELU><<<dim3(NPTS / 64, 4), dim3(128), 0, stream>>>(
            xbf, Wb + lb + LF0, (const float*)d_in[di + 11], nullptr, nullptr,
            hbuf, nullptr, 256, 1024);
        if (l < NLAYERS - 1) {
            gemm_bf16_wmma<EP_RESID><<<dim3(NPTS / 64, 1), dim3(128), 0, stream>>>(
                hbuf, Wb + lb + LF1, (const float*)d_in[di + 13], x32, nullptr,
                xbf, x32, 1024, 256);
        } else {
            gemm_bf16_wmma<EP_RESID_MASK><<<dim3(NPTS / 64, 1), dim3(128), 0, stream>>>(
                hbuf, Wb + lb + LF1, (const float*)d_in[di + 13], x32, mask,
                nullptr, (float*)d_out, 1024, 256);
        }
    }
}
